// MultiHeadAttention_new_48095043781318
// MI455X (gfx1250) — matrix operand order repaired and checked by reading, not compiled, not run on hardware
//
#include <hip/hip_runtime.h>
#include <math.h>

// ---------------- problem constants ----------------
#define BB   2
#define TT   2048
#define DD   512
#define HH   8
#define DKK  64
#define MROWS (BB*TT)        // 4096
#define KTILES (TT/32)       // 64 key tiles of 32

typedef __bf16 bf16;
typedef __attribute__((ext_vector_type(16))) __bf16 v16bf;
typedef __attribute__((ext_vector_type(8)))  __bf16 v8bf;
typedef __attribute__((ext_vector_type(8)))  float  v8f;

// D = A(16x32 bf16) * B(32x16 bf16) + C(16x16 f32)
static __device__ __forceinline__ v8f wmma_bf16(v16bf a, v16bf b, v8f c) {
  return __builtin_amdgcn_wmma_f32_16x16x32_bf16(false, a, false, b,
                                                 (short)0, c, false, false);
}

// --- async global -> LDS copy, 16B per lane (GLOBAL_LOAD_ASYNC_TO_LDS_B128,
//     ASYNCcnt-tracked; see CDNA5 ISA / CDNA5 ISA op 98) ---
static __device__ __forceinline__ void async_ld16(void* ldsPtr, const void* gPtr) {
  // generic pointers to __shared__ carry the LDS offset in their low 32 bits
  unsigned ldsOff = (unsigned)(uintptr_t)ldsPtr;
  asm volatile("global_load_async_to_lds_b128 %0, %1, off"
               :: "v"(ldsOff), "v"(gPtr) : "memory");
}
template <int N>
static __device__ __forceinline__ void wait_async() {
  asm volatile("s_wait_asynccnt %0" :: "i"(N) : "memory");
}

// A fragment from an fp32 row pointer (rowBase points at element k0 of this lane's row)
static __device__ __forceinline__ v16bf a_frag_f32(const float* rowBase, int half) {
  v16bf a;
#pragma unroll
  for (int i = 0; i < 8; ++i) {
    a[i]     = (bf16)rowBase[half * 16 + i];
    a[i + 8] = (bf16)rowBase[half * 16 + 8 + i];
  }
  return a;
}

// A fragment from a bf16 row pointer (16B-aligned chunks)
static __device__ __forceinline__ v16bf a_frag_bf16(const bf16* rowBase, int half) {
  v8bf lo = *reinterpret_cast<const v8bf*>(rowBase + half * 16);
  v8bf hi = *reinterpret_cast<const v8bf*>(rowBase + half * 16 + 8);
  v16bf a;
#pragma unroll
  for (int i = 0; i < 8; ++i) { a[i] = lo[i]; a[i + 8] = hi[i]; }
  return a;
}

// B fragment: 16 contiguous bf16 (32B aligned)
static __device__ __forceinline__ v16bf b_frag_bf16(const bf16* p) {
  return *reinterpret_cast<const v16bf*>(p);
}
// B fragment from 16 contiguous fp32 with conversion
static __device__ __forceinline__ v16bf b_frag_f32(const float* p) {
  v16bf b;
#pragma unroll
  for (int i = 0; i < 16; ++i) b[i] = (bf16)p[i];
  return b;
}

// ======================================================================
// K1: Y[m][n] = X[m][:] . W[n][:] + bias[n]   (X fp32 [4096][512], W fp32 [512][512])
//   transposed==0 : Y[((b*H+h)*T + t)*DK + dk]   (q/k heads)
//   transposed==1 : Y[((b*H+h)*DK + dk)*T + t]   (v heads, for P*V B-frags)
// grid = (8 ntiles of 64, 32 mtiles of 128), block = 256
// ======================================================================
__global__ void __launch_bounds__(256)
mha_proj_qkv_kernel(const float* __restrict__ X, const float* __restrict__ W,
                    const float* __restrict__ bias, bf16* __restrict__ Y,
                    int transposed) {
  const int tid  = threadIdx.x;
  const int w    = tid >> 5;
  const int lane = tid & 31;
  const int half = (lane >> 4) & 1;
  const int n16  = lane & 15;
  const int Nbase = blockIdx.x * 64;
  const int Mbase = blockIdx.y * 128;
  const int arow  = Mbase + w * 16 + n16;

  v8f acc[4];
#pragma unroll
  for (int g = 0; g < 4; ++g)
#pragma unroll
    for (int r = 0; r < 8; ++r) acc[g][r] = 0.0f;

  for (int k0 = 0; k0 < DD; k0 += 32) {
    v16bf a = a_frag_f32(X + (size_t)arow * DD + k0, half);
#pragma unroll
    for (int g = 0; g < 4; ++g) {
      const float* bp = W + (size_t)(Nbase + g * 16 + n16) * DD + k0 + half * 16;
      acc[g] = wmma_bf16(a, b_frag_f32(bp), acc[g]);
    }
  }

#pragma unroll
  for (int g = 0; g < 4; ++g) {
    const int col = Nbase + g * 16 + n16;
    const float bv = bias[col];
    const int h  = col >> 6;
    const int dk = col & 63;
#pragma unroll
    for (int r = 0; r < 8; ++r) {
      const int row = Mbase + w * 16 + r + 8 * half;   // C layout: M = r + 8*half
      const int b   = row >> 11;
      const int t   = row & 2047;
      const float val = acc[g][r] + bv;
      size_t idx = transposed
          ? (((size_t)(b * HH + h) * DKK + dk) * TT + t)
          : (((size_t)(b * HH + h) * TT + t) * DKK + dk);
      Y[idx] = (bf16)val;
    }
  }
}

// ======================================================================
// K2 (fused stats + output): per (b,h, 128-q-row block).
// Pass A: stream K tiles (async double-buffered LDS), online (m, Z).
// Pass B: re-stream K + V tiles, p = (exp(s-m) > Z/T) ? exp(s-m)/Z : 0,
//         O += P*V through per-wave LDS P tile.
// grid = (16 qtiles, 16 bh), block = 256 (8 waves x 16 q rows).
// ======================================================================
__global__ void __launch_bounds__(256)
mha_attn_fused_kernel(const bf16* __restrict__ qh, const bf16* __restrict__ kh,
                      const bf16* __restrict__ vhT, bf16* __restrict__ attnOut) {
  __shared__ __align__(32) bf16 kbuf[2][32 * 64];   // 32 keys x 64 dk
  __shared__ __align__(32) bf16 vbuf[2][64 * 32];   // 64 dk  x 32 keys
  __shared__ __align__(32) bf16 ptile[8][16 * 32];  // per-wave P tile

  const int tid  = threadIdx.x;
  const int w    = tid >> 5;
  const int lane = tid & 31;
  const int half = (lane >> 4) & 1;
  const int n16  = lane & 15;
  const int qt   = blockIdx.x;
  const int bh   = blockIdx.y;
  const int qb   = qt * 128 + w * 16;

  const bf16* qptr = qh  + (size_t)bh * TT * DKK;
  const bf16* kptr = kh  + (size_t)bh * TT * DKK;
  const bf16* vptr = vhT + (size_t)bh * DKK * TT;

  const bf16* qrow = qptr + (size_t)(qb + n16) * DKK;
  v16bf qa0 = a_frag_bf16(qrow, half);        // K = 0..31
  v16bf qa1 = a_frag_bf16(qrow + 32, half);   // K = 32..63

  // K tile = 4KB contiguous (32 consecutive rows of 64 bf16): 16B per thread.
  auto issueK = [&](int buf, int kt) {
    async_ld16(&kbuf[buf][tid * 8], kptr + (size_t)kt * 32 * DKK + tid * 8);
  };
  // V tile = 64 rows (dk) x 32 keys, row stride T: 16B per thread.
  auto issueV = [&](int buf, int kt) {
    const int row = tid >> 2, q4 = tid & 3;
    async_ld16(&vbuf[buf][row * 32 + q4 * 8],
               vptr + (size_t)row * TT + kt * 32 + q4 * 8);
  };

  // ---------------- pass A: online (m, Z) ----------------
  float mrun[8], zrun[8];
#pragma unroll
  for (int r = 0; r < 8; ++r) { mrun[r] = -__builtin_inff(); zrun[r] = 0.0f; }

  issueK(0, 0);
  for (int kt = 0; kt < KTILES; ++kt) {
    const int cur = kt & 1;
    if (kt + 1 < KTILES) { issueK(cur ^ 1, kt + 1); wait_async<1>(); }
    else                 { wait_async<0>(); }
    __syncthreads();

    v8f s[2];
#pragma unroll
    for (int sub = 0; sub < 2; ++sub) {
      const bf16* krow = &kbuf[cur][(sub * 16 + n16) * 64];
#pragma unroll
      for (int r = 0; r < 8; ++r) s[sub][r] = 0.0f;
      s[sub] = wmma_bf16(qa0, b_frag_bf16(krow + half * 16), s[sub]);
      s[sub] = wmma_bf16(qa1, b_frag_bf16(krow + 32 + half * 16), s[sub]);
    }
#pragma unroll
    for (int r = 0; r < 8; ++r) {
      float tmax = fmaxf(s[0][r], s[1][r]);
#pragma unroll
      for (int msk = 1; msk <= 8; msk <<= 1)
        tmax = fmaxf(tmax, __shfl_xor(tmax, msk, 32));
      const float mnew = fmaxf(mrun[r], tmax);
      float e = __expf(s[0][r] - mnew) + __expf(s[1][r] - mnew);
#pragma unroll
      for (int msk = 1; msk <= 8; msk <<= 1)
        e += __shfl_xor(e, msk, 32);
      zrun[r] = zrun[r] * __expf(mrun[r] - mnew) + e;
      mrun[r] = mnew;
    }
    __syncthreads();   // protect kbuf[cur] before next-iteration overwrite
  }

  float invz[8], thr[8];
#pragma unroll
  for (int r = 0; r < 8; ++r) {
    invz[r] = 1.0f / zrun[r];
    thr[r]  = zrun[r] * (1.0f / (float)TT);   // softmax row mean == 1/T
  }

  // ---------------- pass B: O += P * V ----------------
  v8f o[4];
#pragma unroll
  for (int g = 0; g < 4; ++g)
#pragma unroll
    for (int r = 0; r < 8; ++r) o[g][r] = 0.0f;

  issueK(0, 0);
  issueV(0, 0);
  for (int kt = 0; kt < KTILES; ++kt) {
    const int cur = kt & 1;
    if (kt + 1 < KTILES) { issueK(cur ^ 1, kt + 1); issueV(cur ^ 1, kt + 1);
                           wait_async<2>(); }
    else                 { wait_async<0>(); }
    __syncthreads();

#pragma unroll
    for (int sub = 0; sub < 2; ++sub) {
      const bf16* krow = &kbuf[cur][(sub * 16 + n16) * 64];
      v8f s;
#pragma unroll
      for (int r = 0; r < 8; ++r) s[r] = 0.0f;
      s = wmma_bf16(qa0, b_frag_bf16(krow + half * 16), s);
      s = wmma_bf16(qa1, b_frag_bf16(krow + 32 + half * 16), s);
#pragma unroll
      for (int r = 0; r < 8; ++r) {
        const float e = __expf(s[r] - mrun[r]);
        const float p = (e > thr[r]) ? e * invz[r] : 0.0f;
        ptile[w][(r + 8 * half) * 32 + sub * 16 + n16] = (bf16)p;
      }
    }
    __syncthreads();

    v16bf pa = a_frag_bf16(&ptile[w][n16 * 32], half);   // A frag of P (16x32)
#pragma unroll
    for (int g = 0; g < 4; ++g) {
      const bf16* vp = &vbuf[cur][(g * 16 + n16) * 32 + half * 16];
      o[g] = wmma_bf16(pa, b_frag_bf16(vp), o[g]);
    }
    __syncthreads();   // protect kbuf/vbuf[cur] + ptile before overwrite
  }

  const int b = bh >> 3;
  const int h = bh & 7;
#pragma unroll
  for (int g = 0; g < 4; ++g) {
#pragma unroll
    for (int r = 0; r < 8; ++r) {
      const int t = qb + r + 8 * half;
      size_t idx = ((size_t)b * TT + t) * DD + h * DKK + g * 16 + n16;
      attnOut[idx] = (bf16)o[g][r];
    }
  }
}

// ======================================================================
// K3: Out[m][n] = concat[m][:] . Wo[n][:] + bo[n]   (fp32 output)
// grid = (8 ntiles, 32 mtiles), block = 256.
// ======================================================================
__global__ void __launch_bounds__(256)
mha_out_proj_kernel(const bf16* __restrict__ A, const float* __restrict__ Wo,
                    const float* __restrict__ bo, float* __restrict__ Out) {
  const int tid  = threadIdx.x;
  const int w    = tid >> 5;
  const int lane = tid & 31;
  const int half = (lane >> 4) & 1;
  const int n16  = lane & 15;
  const int Nbase = blockIdx.x * 64;
  const int Mbase = blockIdx.y * 128;
  const int arow  = Mbase + w * 16 + n16;

  v8f acc[4];
#pragma unroll
  for (int g = 0; g < 4; ++g)
#pragma unroll
    for (int r = 0; r < 8; ++r) acc[g][r] = 0.0f;

  for (int k0 = 0; k0 < DD; k0 += 32) {
    v16bf a = a_frag_bf16(A + (size_t)arow * DD + k0, half);
#pragma unroll
    for (int g = 0; g < 4; ++g) {
      const float* bp = Wo + (size_t)(Nbase + g * 16 + n16) * DD + k0 + half * 16;
      acc[g] = wmma_bf16(a, b_frag_f32(bp), acc[g]);
    }
  }

#pragma unroll
  for (int g = 0; g < 4; ++g) {
    const int col = Nbase + g * 16 + n16;
    const float bv = bo[col];
#pragma unroll
    for (int r = 0; r < 8; ++r) {
      const int row = Mbase + w * 16 + r + 8 * half;
      Out[(size_t)row * DD + col] = acc[g][r] + bv;
    }
  }
}

// ======================================================================
extern "C" void kernel_launch(void* const* d_in, const int* in_sizes, int n_in,
                              void* d_out, int out_size, void* d_ws, size_t ws_size,
                              hipStream_t stream) {
  const float* q  = (const float*)d_in[0];
  const float* k  = (const float*)d_in[1];
  const float* v  = (const float*)d_in[2];
  const float* Wq = (const float*)d_in[3];
  const float* bq = (const float*)d_in[4];
  const float* Wk = (const float*)d_in[5];
  const float* bk = (const float*)d_in[6];
  const float* Wv = (const float*)d_in[7];
  const float* bv = (const float*)d_in[8];
  const float* Wo = (const float*)d_in[9];
  const float* bo = (const float*)d_in[10];

  char* ws = (char*)d_ws;
  const size_t headElems = (size_t)BB * HH * TT * DKK;   // 2,097,152
  bf16* qh   = (bf16*)(ws + 0);
  bf16* kh   = (bf16*)(ws + headElems * 2);
  bf16* vhT  = (bf16*)(ws + headElems * 4);
  bf16* attn = (bf16*)(ws + headElems * 6);

  dim3 blk(256);
  dim3 gProj(DD / 64, MROWS / 128);   // (8, 32)
  dim3 gAttn(TT / 128, BB * HH);      // (16, 16)

  mha_proj_qkv_kernel<<<gProj, blk, 0, stream>>>(q, Wq, bq, qh, 0);
  mha_proj_qkv_kernel<<<gProj, blk, 0, stream>>>(k, Wk, bk, kh, 0);
  mha_proj_qkv_kernel<<<gProj, blk, 0, stream>>>(v, Wv, bv, vhT, 1);
  mha_attn_fused_kernel<<<gAttn, blk, 0, stream>>>(qh, kh, vhT, attn);
  mha_out_proj_kernel<<<gProj, blk, 0, stream>>>(attn, Wo, bo, (float*)d_out);
}
